// EncoderBlock_66331474919574
// MI455X (gfx1250) — matrix operand order repaired and checked by reading, not compiled, not run on hardware
//
#include <hip/hip_runtime.h>
#include <hip/hip_bf16.h>
#include <math.h>

// ---------------------------------------------------------------------------
// Encoder block for gfx1250 (MI455X, wave32, WMMA).
//   B=4, S=1024, D=1024, H=16, DV=64, DFF=4096, M=B*S=4096
// Compute-bound (~120 GFLOP vs ~110MB traffic) -> everything through
// v_wmma_f32_16x16x32_bf16 with LDS-staged tiles; flash-style attention.
// ---------------------------------------------------------------------------

#define Bn   4
#define Sn   1024
#define Dn   1024
#define Hn   16
#define DVn  64
#define DFFn 4096
#define Mn   (Bn * Sn)

typedef __attribute__((ext_vector_type(16))) __bf16 v16bf;
typedef __attribute__((ext_vector_type(8)))  __bf16 v8bf;
typedef __attribute__((ext_vector_type(2)))  __bf16 v2bf;
typedef __attribute__((ext_vector_type(8)))  float  v8f;

#define WMMA_BF16(a, b, c)                                                    \
  __builtin_amdgcn_wmma_f32_16x16x32_bf16(false, (a), false, (b), (short)0,   \
                                          (c), false, false)

// A-matrix fragment (16x32 bf16, M x K), ISA 7.12.2 layout:
// lanes 0-15: M=lane; lanes 16-31: M=lane-16 with K offset +8 within group.
// VGPR v: K pair = (v>>2)*16 + half*8 + (v&3)*2.
__device__ __forceinline__ v16bf load_a_frag(const __bf16* __restrict__ base,
                                             int ld, int lane) {
  const int m  = lane & 15;
  const int hh = lane >> 4;
  const __bf16* row = base + (size_t)m * ld;
  v16bf r;
#pragma unroll
  for (int v = 0; v < 8; ++v) {
    const int k = ((v >> 2) << 4) + (hh << 3) + ((v & 3) << 1);
    v2bf p = *(const v2bf*)(row + k);
    r[2 * v]     = p.x;
    r[2 * v + 1] = p.y;
  }
  return r;
}

// B-matrix fragment (32x16 bf16, K x N): N = lane&15, same K layout as A:
// elements 0-7 hold K = half*8 + 0..7, elements 8-15 hold K = 16 + half*8 + 0..7,
// read from a row-major [N][K] tile (W is stored [N,K], C = A @ W^T).
__device__ __forceinline__ v16bf load_b_frag(const __bf16* __restrict__ base,
                                             int ld, int lane) {
  const __bf16* p = base + (size_t)(lane & 15) * ld + ((lane >> 4) << 3);
  v8bf lo = *(const v8bf*)p;
  v8bf hi = *(const v8bf*)(p + 16);
  return __builtin_shufflevector(lo, hi, 0, 1, 2, 3, 4, 5, 6, 7, 8, 9, 10, 11,
                                 12, 13, 14, 15);
}

// ---------------------------------------------------------------------------
// fp32 -> bf16 conversion (weights / once per launch)
// ---------------------------------------------------------------------------
__global__ void cvt_bf16_kernel(const float* __restrict__ src,
                                __bf16* __restrict__ dst, int n) {
  int i = (blockIdx.x * 256 + threadIdx.x) * 4;
  if (i + 3 < n) {
#pragma unroll
    for (int e = 0; e < 4; ++e) dst[i + e] = (__bf16)src[i + e];
  }
}

// ---------------------------------------------------------------------------
// LayerNorm (torch-style unbiased std, repo eps = -1e6), fp32 in -> bf16 out
// one 256-thread block per row of D=1024
// ---------------------------------------------------------------------------
__global__ void ln_bf16_kernel(const float* __restrict__ x,
                               const float* __restrict__ alpha,
                               const float* __restrict__ bias,
                               __bf16* __restrict__ out) {
  __shared__ float s1[256], s2[256];
  const int row = blockIdx.x, t = threadIdx.x;
  const float* xr = x + (size_t)row * Dn;
  float v[4], sum = 0.f, sq = 0.f;
#pragma unroll
  for (int e = 0; e < 4; ++e) {
    v[e] = xr[t * 4 + e];
    sum += v[e];
    sq += v[e] * v[e];
  }
  s1[t] = sum;
  s2[t] = sq;
  __syncthreads();
  for (int off = 128; off > 0; off >>= 1) {
    if (t < off) {
      s1[t] += s1[t + off];
      s2[t] += s2[t + off];
    }
    __syncthreads();
  }
  const float mean = s1[0] * (1.0f / Dn);
  const float var  = (s2[0] - (float)Dn * mean * mean) * (1.0f / (Dn - 1));
  const float den  = sqrtf(fmaxf(var, 0.f)) + (-1000000.0f);  // repo eps
  const float a = alpha[0], bb = bias[0];
  __bf16* orow = out + (size_t)row * Dn;
#pragma unroll
  for (int e = 0; e < 4; ++e)
    orow[t * 4 + e] = (__bf16)(a * (v[e] - mean) / den + bb);
}

// ---------------------------------------------------------------------------
// GEMM: C[M,N] = A[M,K](bf16) @ W[N,K]^T(bf16) + bias (+residual) (+relu)
// 128x128 block tile, 8 waves (4x2), each wave 32x64 = 2x4 WMMA frags.
// ---------------------------------------------------------------------------
__global__ void __launch_bounds__(256)
gemm_nt_kernel(const __bf16* __restrict__ A, const __bf16* __restrict__ W,
               const float* __restrict__ bias,
               const float* __restrict__ residual, float* __restrict__ Cf,
               __bf16* __restrict__ Cb, int Md, int Nd, int Kd, int relu) {
  __shared__ __align__(16) __bf16 As[128 * 32];
  __shared__ __align__(16) __bf16 Bs[128 * 32];

  const int lane = threadIdx.x & 31, wave = threadIdx.x >> 5;
  const int wm = wave >> 1, wn = wave & 1;
  const int mtile = blockIdx.y * 128, ntile = blockIdx.x * 128;

  v8f c[2][4] = {};
  const int r  = threadIdx.x >> 1;
  const int cb = (threadIdx.x & 1) * 16;

  for (int kt = 0; kt < Kd; kt += 32) {
    const __bf16* ga = A + (size_t)(mtile + r) * Kd + kt + cb;
    const __bf16* gb = W + (size_t)(ntile + r) * Kd + kt + cb;
    *(v8bf*)&As[r * 32 + cb]     = *(const v8bf*)ga;
    *(v8bf*)&As[r * 32 + cb + 8] = *(const v8bf*)(ga + 8);
    *(v8bf*)&Bs[r * 32 + cb]     = *(const v8bf*)gb;
    *(v8bf*)&Bs[r * 32 + cb + 8] = *(const v8bf*)(gb + 8);
    if (kt + 32 < Kd) {  // hint next K tile -> global_prefetch_b8
      __builtin_prefetch(ga + 32, 0, 1);
      __builtin_prefetch(gb + 32, 0, 1);
    }
    __syncthreads();

    v16bf a0 = load_a_frag(&As[(wm * 32) * 32], 32, lane);
    v16bf a1 = load_a_frag(&As[(wm * 32 + 16) * 32], 32, lane);
#pragma unroll
    for (int j = 0; j < 4; ++j) {
      v16bf bf = load_b_frag(&Bs[(wn * 64 + j * 16) * 32], 32, lane);
      c[0][j] = WMMA_BF16(a0, bf, c[0][j]);
      c[1][j] = WMMA_BF16(a1, bf, c[1][j]);
    }
    __syncthreads();
  }

  const int n = lane & 15, hh = lane >> 4;
#pragma unroll
  for (int i = 0; i < 2; ++i) {
#pragma unroll
    for (int j = 0; j < 4; ++j) {
#pragma unroll
      for (int v = 0; v < 8; ++v) {
        const int m  = mtile + wm * 32 + i * 16 + hh * 8 + v;
        const int nc = ntile + wn * 64 + j * 16 + n;
        float val = c[i][j][v] + bias[nc];
        if (relu) val = fmaxf(val, 0.f);
        if (residual) val += residual[(size_t)m * Nd + nc];
        if (Cf) Cf[(size_t)m * Nd + nc] = val;
        if (Cb) Cb[(size_t)m * Nd + nc] = (__bf16)val;
      }
    }
  }
}

// ---------------------------------------------------------------------------
// Flash attention: grid = B*H*(S/128); 8 waves, each owns 16 query rows.
// KV tiles of 32 staged in LDS (V transposed), online softmax, WMMA for
// both Q@K^T and P@V. scale = 1/sqrt(64), mask==0 -> -1e9.
// ---------------------------------------------------------------------------
__global__ void __launch_bounds__(256)
attn_kernel(const __bf16* __restrict__ Q, const __bf16* __restrict__ K,
            const __bf16* __restrict__ V, const int* __restrict__ mask,
            __bf16* __restrict__ O) {
  __shared__ __align__(16) __bf16 Ks[32 * 64];   // [kv][d]
  __shared__ __align__(16) __bf16 Vst[64 * 32];  // [d][kv] (transposed)
  __shared__ __align__(16) __bf16 Ps[8 * 16 * 32];
  __shared__ int ms[32];

  const int lane = threadIdx.x & 31, wave = threadIdx.x >> 5;
  const int qc = blockIdx.x & 7;        // S/128 = 8 chunks
  const int bh = blockIdx.x >> 3;
  const int b = bh >> 4, h = bh & 15;   // H = 16
  const int q0 = qc * 128 + wave * 16;
  const int n = lane & 15, hh = lane >> 4;

  const __bf16* qbase = Q + ((size_t)(b * Sn + q0)) * Dn + h * DVn;
  v16bf aq0 = load_a_frag(qbase, Dn, lane);
  v16bf aq1 = load_a_frag(qbase + 32, Dn, lane);

  float rmax[8], rsum[8];
  v8f co[4] = {};
#pragma unroll
  for (int v = 0; v < 8; ++v) {
    rmax[v] = -1e30f;
    rsum[v] = 0.f;
  }

  const int kvr = threadIdx.x >> 3;
  const int dcb = (threadIdx.x & 7) * 8;

  for (int t = 0; t < Sn / 32; ++t) {
    const int kv0 = t * 32;
    const size_t grow = (size_t)(b * Sn + kv0 + kvr) * Dn + h * DVn + dcb;
    *(v8bf*)&Ks[kvr * 64 + dcb] = *(const v8bf*)(K + grow);
    v8bf vv = *(const v8bf*)(V + grow);
#pragma unroll
    for (int e = 0; e < 8; ++e) Vst[(dcb + e) * 32 + kvr] = vv[e];
    if (threadIdx.x < 32) ms[threadIdx.x] = mask[b * Sn + kv0 + threadIdx.x];
    __syncthreads();

    // scores: 16(q) x 32(kv), K-dim = 64
    v8f s0 = {}, s1 = {};
    v16bf bk;
    bk = load_b_frag(&Ks[0 * 64 + 0], 64, lane);
    s0 = WMMA_BF16(aq0, bk, s0);
    bk = load_b_frag(&Ks[0 * 64 + 32], 64, lane);
    s0 = WMMA_BF16(aq1, bk, s0);
    bk = load_b_frag(&Ks[16 * 64 + 0], 64, lane);
    s1 = WMMA_BF16(aq0, bk, s1);
    bk = load_b_frag(&Ks[16 * 64 + 32], 64, lane);
    s1 = WMMA_BF16(aq1, bk, s1);

    const int m0 = ms[n], m1 = ms[16 + n];
#pragma unroll
    for (int v = 0; v < 8; ++v) {
      float sv0 = m0 ? s0[v] * 0.125f : -1e9f;
      float sv1 = m1 ? s1[v] * 0.125f : -1e9f;
      float tv = fmaxf(sv0, sv1);
      tv = fmaxf(tv, __shfl_xor(tv, 1, 32));
      tv = fmaxf(tv, __shfl_xor(tv, 2, 32));
      tv = fmaxf(tv, __shfl_xor(tv, 4, 32));
      tv = fmaxf(tv, __shfl_xor(tv, 8, 32));
      const float mnew = fmaxf(rmax[v], tv);
      const float sf = __expf(rmax[v] - mnew);
      const float p0 = __expf(sv0 - mnew);
      const float p1 = __expf(sv1 - mnew);
      float ps = p0 + p1;
      ps += __shfl_xor(ps, 1, 32);
      ps += __shfl_xor(ps, 2, 32);
      ps += __shfl_xor(ps, 4, 32);
      ps += __shfl_xor(ps, 8, 32);
      rsum[v] = rsum[v] * sf + ps;
      rmax[v] = mnew;
#pragma unroll
      for (int j = 0; j < 4; ++j) co[j][v] *= sf;
      const int prow = hh * 8 + v;
      Ps[wave * 512 + prow * 32 + n]      = (__bf16)p0;
      Ps[wave * 512 + prow * 32 + 16 + n] = (__bf16)p1;
    }

    // ctx += P(16x32) @ V(32x64)
    v16bf pf = load_a_frag(&Ps[wave * 512], 32, lane);
#pragma unroll
    for (int j = 0; j < 4; ++j) {
      v16bf bv = load_b_frag(&Vst[(j * 16) * 32], 32, lane);
      co[j] = WMMA_BF16(pf, bv, co[j]);
    }
    __syncthreads();
  }

#pragma unroll
  for (int v = 0; v < 8; ++v) {
    const float inv = 1.0f / rsum[v];
    const size_t row = (size_t)(b * Sn + q0 + hh * 8 + v);
#pragma unroll
    for (int j = 0; j < 4; ++j)
      O[row * Dn + h * DVn + j * 16 + n] = (__bf16)(co[j][v] * inv);
  }
}

// ---------------------------------------------------------------------------
extern "C" void kernel_launch(void* const* d_in, const int* in_sizes, int n_in,
                              void* d_out, int out_size, void* d_ws,
                              size_t ws_size, hipStream_t stream) {
  const float* x    = (const float*)d_in[0];
  const int*   msk  = (const int*)d_in[1];
  const float* wq = (const float*)d_in[2];
  const float* bq = (const float*)d_in[3];
  const float* wk = (const float*)d_in[4];
  const float* bk = (const float*)d_in[5];
  const float* wv = (const float*)d_in[6];
  const float* bv = (const float*)d_in[7];
  const float* wo = (const float*)d_in[8];
  const float* bo = (const float*)d_in[9];
  const float* w1 = (const float*)d_in[10];
  const float* b1 = (const float*)d_in[11];
  const float* w2 = (const float*)d_in[12];
  const float* b2 = (const float*)d_in[13];
  const float* alpha1 = (const float*)d_in[14];
  const float* bias1  = (const float*)d_in[15];
  const float* alpha2 = (const float*)d_in[16];
  const float* bias2  = (const float*)d_in[17];
  float* out = (float*)d_out;

  // workspace carve-up (bf16 scratch + f32 residual stream)
  char* w = (char*)d_ws;
  auto take = [&](size_t bytes) {
    char* p = w;
    w += (bytes + 255) & ~(size_t)255;
    return p;
  };
  __bf16* wqb = (__bf16*)take((size_t)Dn * Dn * 2);
  __bf16* wkb = (__bf16*)take((size_t)Dn * Dn * 2);
  __bf16* wvb = (__bf16*)take((size_t)Dn * Dn * 2);
  __bf16* wob = (__bf16*)take((size_t)Dn * Dn * 2);
  __bf16* w1b = (__bf16*)take((size_t)DFFn * Dn * 2);
  __bf16* w2b = (__bf16*)take((size_t)Dn * DFFn * 2);
  __bf16* xn1 = (__bf16*)take((size_t)Mn * Dn * 2);
  __bf16* qb  = (__bf16*)take((size_t)Mn * Dn * 2);
  __bf16* kb  = (__bf16*)take((size_t)Mn * Dn * 2);
  __bf16* vb  = (__bf16*)take((size_t)Mn * Dn * 2);
  __bf16* ctx = (__bf16*)take((size_t)Mn * Dn * 2);
  float*  x2  = (float*)take((size_t)Mn * Dn * 4);
  __bf16* xn2 = (__bf16*)take((size_t)Mn * Dn * 2);
  __bf16* hb  = (__bf16*)take((size_t)Mn * DFFn * 2);

  const dim3 blk(256);

  // 0) weights -> bf16
  cvt_bf16_kernel<<<(Dn * Dn) / 1024, blk, 0, stream>>>(wq, wqb, Dn * Dn);
  cvt_bf16_kernel<<<(Dn * Dn) / 1024, blk, 0, stream>>>(wk, wkb, Dn * Dn);
  cvt_bf16_kernel<<<(Dn * Dn) / 1024, blk, 0, stream>>>(wv, wvb, Dn * Dn);
  cvt_bf16_kernel<<<(Dn * Dn) / 1024, blk, 0, stream>>>(wo, wob, Dn * Dn);
  cvt_bf16_kernel<<<(DFFn * Dn) / 1024, blk, 0, stream>>>(w1, w1b, DFFn * Dn);
  cvt_bf16_kernel<<<(Dn * DFFn) / 1024, blk, 0, stream>>>(w2, w2b, Dn * DFFn);

  // 1) LN1
  ln_bf16_kernel<<<Mn, blk, 0, stream>>>(x, alpha1, bias1, xn1);

  // 2) Q/K/V projections
  dim3 gD(Dn / 128, Mn / 128);
  gemm_nt_kernel<<<gD, blk, 0, stream>>>(xn1, wqb, bq, nullptr, nullptr, qb,
                                         Mn, Dn, Dn, 0);
  gemm_nt_kernel<<<gD, blk, 0, stream>>>(xn1, wkb, bk, nullptr, nullptr, kb,
                                         Mn, Dn, Dn, 0);
  gemm_nt_kernel<<<gD, blk, 0, stream>>>(xn1, wvb, bv, nullptr, nullptr, vb,
                                         Mn, Dn, Dn, 0);

  // 3) attention
  attn_kernel<<<Bn * Hn * (Sn / 128), blk, 0, stream>>>(qb, kb, vb, msk, ctx);

  // 4) output projection + residual (x2 = x + ctx @ wo^T + bo)
  gemm_nt_kernel<<<gD, blk, 0, stream>>>(ctx, wob, bo, x, x2, nullptr, Mn, Dn,
                                         Dn, 0);

  // 5) LN2
  ln_bf16_kernel<<<Mn, blk, 0, stream>>>(x2, alpha2, bias2, xn2);

  // 6) FFN up + ReLU
  dim3 gF(DFFn / 128, Mn / 128);
  gemm_nt_kernel<<<gF, blk, 0, stream>>>(xn2, w1b, b1, nullptr, nullptr, hb,
                                         Mn, DFFn, Dn, 1);

  // 7) FFN down + residual -> out
  gemm_nt_kernel<<<gD, blk, 0, stream>>>(hb, w2b, b2, x2, out, nullptr, Mn, Dn,
                                         DFFn, 0);
}
